// AllegroConditioner_41326175322517
// MI455X (gfx1250) — compile-verified, matrix operand order checked by reading, not run on hardware
//
#include <hip/hip_runtime.h>

typedef __attribute__((ext_vector_type(16))) _Float16 v16h;
typedef __attribute__((ext_vector_type(8)))  float    v8f;
typedef __attribute__((ext_vector_type(4)))  float    f32x4;
typedef __attribute__((ext_vector_type(4)))  unsigned int v4u;
typedef __attribute__((ext_vector_type(4)))  int      v4i;
typedef __attribute__((ext_vector_type(8)))  int      v8i;

#define BATCH   4096
#define REST    160
#define ZDIM    2208      // REST + 32*64
#define HID0    512
#define HID1    512
#define DOUT    256
#define PI_F    3.14159265358979323846f

#if defined(__gfx1250__) && __has_builtin(__builtin_amdgcn_tensor_load_to_lds)
#define USE_TDM 1
#else
#define USE_TDM 0
#endif

// ---------------------------------------------------------------------------
// Load a 16-half WMMA fragment: two contiguous 8-half (16B) chunks.
// A 16x32 f16 layout (ISA 7.12.2): lane<16 -> K {0..7},{16..23};
// lane>=16 -> {8..15},{24..31}. Same pattern used for B (col-major staged).
// ---------------------------------------------------------------------------
__device__ __forceinline__ v16h ld_frag16(const _Float16* base, int o0, int o1) {
  union { v16h v; f32x4 f[2]; } u;
  u.f[0] = *reinterpret_cast<const f32x4*>(base + o0);
  u.f[1] = *reinterpret_cast<const f32x4*>(base + o1);
  return u.v;
}

#if USE_TDM
// ---------------------------------------------------------------------------
// TDM 2D tile load: rows x 32 halves, row stride K halves, into LDS (packed).
// D# per CDNA5 ISA 8.3/8.4: group0 = {flags, lds_addr, global_addr, type=2},
// group1 = {data_size=2B, tensor_dim0=K, tensor_dim1=tensor_rows, tile=32 x rows,
//           dim0_stride=K}. Groups 2/3 zero (2D tensor).
// ---------------------------------------------------------------------------
__device__ __forceinline__ void tdm_load_2d(unsigned lds_addr,
                                            unsigned long long gaddr,
                                            int K, int rows, int tensor_rows) {
  v4u g0;
  g0.x = 1u;                                              // count=1, user mode
  g0.y = lds_addr;                                        // LDS byte address
  g0.z = (unsigned)(gaddr & 0xffffffffull);               // global_addr lo
  g0.w = (unsigned)((gaddr >> 32) & 0x01ffffffull)        // global_addr hi
         | (2u << 30);                                    // type = 2 (image)
  v8i g1;
  g1[0] = (int)(1u << 16);                                // data_size = 2 bytes
  g1[1] = (int)(((unsigned)K & 0xffffu) << 16);           // tensor_dim0 lo16
  g1[2] = (int)(((unsigned)K >> 16) |
                (((unsigned)tensor_rows & 0xffffu) << 16)); // dim0 hi / dim1 lo
  g1[3] = (int)(((unsigned)tensor_rows >> 16) | (32u << 16)); // tile_dim0 = 32
  g1[4] = rows;                                           // tile_dim1, tile_dim2=0
  g1[5] = K;                                              // tensor_dim0_stride lo
  g1[6] = 0;                                              // stride hi / dim1_stride
  g1[7] = 0;
  v4i z4 = 0;
#if __clang_major__ >= 23
  v8i z8 = 0;
  __builtin_amdgcn_tensor_load_to_lds(g0, g1, z4, z4, z8, 0);
#else
  __builtin_amdgcn_tensor_load_to_lds(g0, g1, z4, z4, 0);
#endif
}
#endif

// ---------------------------------------------------------------------------
// Weight convert: w[K][N] fp32 -> wt[N][K] f16 (contiguous K per column)
// ---------------------------------------------------------------------------
__global__ void transpose_to_f16(const float* __restrict__ w, _Float16* __restrict__ wt,
                                 int K, int N) {
  int idx = blockIdx.x * 256 + threadIdx.x;
  if (idx >= K * N) return;
  int n = idx / K, k = idx - n * K;
  wt[idx] = (_Float16)w[(size_t)k * N + n];
}

// ---------------------------------------------------------------------------
// GNN: per-sample geometry -> RBF -> silu MLP -> masked neighbor sum -> w2 GEMM
// One block per sample, 256 threads = 8 waves (wave32).
// Writes z (f16): [x_rest(160) | atom(32*64)]
// ---------------------------------------------------------------------------
__global__ void gnn_kernel(const float* __restrict__ x,
                           const float* __restrict__ w1, const float* __restrict__ b1,
                           const float* __restrict__ w2, const float* __restrict__ b2,
                           _Float16* __restrict__ z) {
  __shared__ float s_cart[96];
  __shared__ float s_w1[8 * 64];
  __shared__ float s_b1[64];
  __shared__ float s_b2[64];
  __shared__ alignas(16) _Float16 s_w2t[64 * 64];   // [n][k]
  __shared__ float s_feat[1024 * 8];                // [pair][basis]
  __shared__ unsigned char s_mask[1024];
  __shared__ alignas(16) _Float16 s_atom[32 * 64];  // masked-summed silu feats (f16)
  __shared__ float s_nb[32];

  const int b   = blockIdx.x;
  const int tid = threadIdx.x;
  const float* xb = x + (size_t)b * 256;

  if (tid < 96) s_cart[tid] = xb[REST + tid];
  for (int i = tid; i < 512; i += 256) s_w1[i] = w1[i];
  if (tid < 64) { s_b1[tid] = b1[tid]; s_b2[tid] = b2[tid]; }
  for (int i = tid; i < 4096; i += 256) {
    int k = i >> 6, n = i & 63;
    s_w2t[n * 64 + k] = (_Float16)w2[i];            // w2 is [k][n] row-major
  }
  __syncthreads();

  // --- pairwise distances + masked RBF * envelope features ---
  for (int p = tid; p < 1024; p += 256) {
    int i = p >> 5, j = p & 31;
    float dx = s_cart[i * 3 + 0] - s_cart[j * 3 + 0];
    float dy = s_cart[i * 3 + 1] - s_cart[j * 3 + 1];
    float dz = s_cart[i * 3 + 2] - s_cart[j * 3 + 2];
    float d  = sqrtf(dx * dx + dy * dy + dz * dz + 1e-12f);
    bool  m  = (d <= 5.0f) && (i != j);
    s_mask[p] = (unsigned char)m;
    float ds  = m ? d : 1.0f;
    float u   = ds * 0.2f;
    float env = 1.0f + u * u * u * (-10.0f + u * (15.0f - 6.0f * u));
    float inv = env / ds;
    float ang = PI_F * 0.2f * ds;
#pragma unroll
    for (int k = 0; k < 8; k++)
      s_feat[p * 8 + k] = __sinf(ang * (float)(k + 1)) * inv;
  }
  __syncthreads();

  // --- silu(feat@w1 + b1), masked sum over neighbors j ---
  {
    int i  = tid >> 3;          // atom 0..31
    int cb = (tid & 7) * 8;     // 8 latent channels per thread
    float acc[8];
#pragma unroll
    for (int c = 0; c < 8; c++) acc[c] = 0.0f;
    int cnt = 0;
    for (int j = 0; j < 32; j++) {
      if (!s_mask[i * 32 + j]) continue;
      cnt++;
      const float* f = &s_feat[(i * 32 + j) * 8];
#pragma unroll
      for (int c = 0; c < 8; c++) {
        float pre = s_b1[cb + c];
#pragma unroll
        for (int k = 0; k < 8; k++) pre += f[k] * s_w1[k * 64 + cb + c];
        acc[c] += pre / (1.0f + __expf(-pre));      // silu
      }
    }
#pragma unroll
    for (int c = 0; c < 8; c++) s_atom[i * 64 + cb + c] = (_Float16)acc[c];
    if ((tid & 7) == 0) s_nb[i] = (float)cnt;
  }
  __syncthreads();

  // --- atom(32x64) @ w2(64x64) + n_i * b2 via WMMA f16 ---
  {
    const int lane = tid & 31, wave = tid >> 5;
    const int mt = wave & 1;          // 2 M-tiles of 16
    const int nt = wave >> 1;         // 4 N-tiles of 16
    const int r  = lane & 15;
    const int hi = lane >> 4;
    v8f c = {};
#pragma unroll
    for (int kk = 0; kk < 64; kk += 32) {
      int o0 = (hi ? 8 : 0) + kk, o1 = (hi ? 24 : 16) + kk;
      v16h a = ld_frag16(&s_atom[(mt * 16 + r) * 64], o0, o1);
      v16h w = ld_frag16(&s_w2t[(nt * 16 + r) * 64], o0, o1);
      c = __builtin_amdgcn_wmma_f32_16x16x32_f16(false, a, false, w, (short)0, c,
                                                 false, false);
    }
    _Float16* zb = z + (size_t)b * ZDIM + REST;
#pragma unroll
    for (int q = 0; q < 8; q++) {
      int m = mt * 16 + q + hi * 8;
      int n = nt * 16 + r;
      float v = c[q] + s_nb[m] * s_b2[n];
      zb[m * 64 + n] = (_Float16)v;
    }
  }

  if (tid < REST) z[(size_t)b * ZDIM + tid] = (_Float16)xb[tid];
}

// ---------------------------------------------------------------------------
// Tiled WMMA GEMM: out = act(A(MxK,f16) @ W(KxN) + bias), W staged as wt[N][K].
// Block tile 128x64, BK=32, 256 threads = 8 waves; wave owns a 16x64 strip
// (1 A-frag + 4 WMMAs per K-step). Double-buffered LDS; one barrier per
// K-step. Staging via Tensor Data Mover (wave 0 issues tensor_load_to_lds,
// completion via s_wait_tensorcnt) when available; else direct b128 copies.
// MODE 0: tanh epilogue -> f16; MODE 1: bias only -> f32.
// ---------------------------------------------------------------------------
template <int MODE>
__global__ void gemm_wmma(const _Float16* __restrict__ A,
                          const _Float16* __restrict__ Wt,
                          const float* __restrict__ bias,
                          void* __restrict__ outv,
                          int M, int N, int K) {
  __shared__ alignas(16) _Float16 sA[2][128 * 32];   // [m][k] per buffer (8KB)
  __shared__ alignas(16) _Float16 sW[2][64 * 32];    // [n][k] per buffer (4KB)

  const int m0 = blockIdx.x * 128;
  const int n0 = blockIdx.y * 64;
  const int tid  = threadIdx.x;
  const int lane = tid & 31, wave = tid >> 5;
  const int mt = wave;                      // 8 M-tiles of 16 rows
  const int r  = lane & 15;
  const int hi = lane >> 4;
  const int o0 = hi ? 8 : 0, o1 = hi ? 24 : 16;

  v8f acc[4] = {{}, {}, {}, {}};

#if USE_TDM
  const unsigned long long aBase =
      (unsigned long long)(size_t)(const void*)(A + (size_t)m0 * K);
  const unsigned long long wBase =
      (unsigned long long)(size_t)(const void*)(Wt + (size_t)n0 * K);
  const unsigned ldsA = (unsigned)(size_t)(const void*)&sA[0][0];
  const unsigned ldsW = (unsigned)(size_t)(const void*)&sW[0][0];
#endif

  auto stage = [&](int buf, int k0) {
#if USE_TDM
    if (wave == 0) {
      tdm_load_2d(ldsA + (unsigned)buf * (128 * 32 * 2),
                  aBase + (unsigned long long)k0 * 2, K, 128, M);
      tdm_load_2d(ldsW + (unsigned)buf * (64 * 32 * 2),
                  wBase + (unsigned long long)k0 * 2, K, 64, N);
    }
#else
    // A tile: 128 rows x 4 chunks; W tile: 64 rows x 4 chunks (16B each)
#pragma unroll
    for (int idx = tid; idx < 512; idx += 256) {
      int row = idx >> 2, ch = (idx & 3) * 8;
      *reinterpret_cast<f32x4*>(&sA[buf][row * 32 + ch]) =
          *reinterpret_cast<const f32x4*>(&A[(size_t)(m0 + row) * K + k0 + ch]);
    }
    {
      int row = tid >> 2, ch = (tid & 3) * 8;
      *reinterpret_cast<f32x4*>(&sW[buf][row * 32 + ch]) =
          *reinterpret_cast<const f32x4*>(&Wt[(size_t)(n0 + row) * K + k0 + ch]);
    }
#endif
  };

  stage(0, 0);
#if USE_TDM
  if (wave == 0) __builtin_amdgcn_s_wait_tensorcnt(0);
#endif
  __syncthreads();

  const int nk = K >> 5;
  for (int i = 0; i < nk; ++i) {
    const int cur = i & 1;
    if (i + 1 < nk) stage(cur ^ 1, (i + 1) << 5);   // async fill of other buffer

    v16h a = ld_frag16(&sA[cur][(mt * 16 + r) * 32], o0, o1);
#pragma unroll
    for (int nt = 0; nt < 4; nt++) {
      v16h w = ld_frag16(&sW[cur][(nt * 16 + r) * 32], o0, o1);
      acc[nt] = __builtin_amdgcn_wmma_f32_16x16x32_f16(false, a, false, w, (short)0,
                                                       acc[nt], false, false);
    }

#if USE_TDM
    if (wave == 0) __builtin_amdgcn_s_wait_tensorcnt(0);
#endif
    __syncthreads();
  }

  // epilogue: C layout -> VGPR q: M = q + 8*hi ; N = lane%16
#pragma unroll
  for (int nt = 0; nt < 4; nt++) {
#pragma unroll
    for (int q = 0; q < 8; q++) {
      int m = m0 + mt * 16 + q + hi * 8;
      int n = n0 + nt * 16 + r;
      float v = acc[nt][q] + bias[n];
      if (MODE == 0) {
        ((_Float16*)outv)[(size_t)m * N + n] = (_Float16)tanhf(v);
      } else {
        ((float*)outv)[(size_t)m * N + n] = v;
      }
    }
  }
}

// ---------------------------------------------------------------------------
// Host launch
// ---------------------------------------------------------------------------
extern "C" void kernel_launch(void* const* d_in, const int* in_sizes, int n_in,
                              void* d_out, int out_size, void* d_ws, size_t ws_size,
                              hipStream_t stream) {
  const float* x   = (const float*)d_in[0];
  const float* w1  = (const float*)d_in[1];
  const float* b1  = (const float*)d_in[2];
  const float* w2  = (const float*)d_in[3];
  const float* b2  = (const float*)d_in[4];
  const float* wd0 = (const float*)d_in[5];
  const float* bd0 = (const float*)d_in[6];
  const float* wd1 = (const float*)d_in[7];
  const float* bd1 = (const float*)d_in[8];
  const float* wd2 = (const float*)d_in[9];
  const float* bd2 = (const float*)d_in[10];

  char* ws = (char*)d_ws;
  _Float16* z   = (_Float16*)(ws + 0);                    // 4096*2208*2 = 18,087,936
  _Float16* z1  = (_Float16*)(ws + 18087936);             // 4096*512*2  =  4,194,304
  _Float16* z2  = (_Float16*)(ws + 22282240);             // 4096*512*2  =  4,194,304
  _Float16* wt0 = (_Float16*)(ws + 26476544);             // 512*2208*2  =  2,260,992
  _Float16* wt1 = (_Float16*)(ws + 28737536);             // 512*512*2   =    524,288
  _Float16* wt2 = (_Float16*)(ws + 29261824);             // 256*512*2   =    262,144

  // weight converts (transposed to [N][K] f16)
  transpose_to_f16<<<(ZDIM * HID0 + 255) / 256, 256, 0, stream>>>(wd0, wt0, ZDIM, HID0);
  transpose_to_f16<<<(HID0 * HID1 + 255) / 256, 256, 0, stream>>>(wd1, wt1, HID0, HID1);
  transpose_to_f16<<<(HID1 * DOUT + 255) / 256, 256, 0, stream>>>(wd2, wt2, HID1, DOUT);

  // GNN: one block per sample
  gnn_kernel<<<BATCH, 256, 0, stream>>>(x, w1, b1, w2, b2, z);

  // dense head
  gemm_wmma<0><<<dim3(BATCH / 128, HID0 / 64), 256, 0, stream>>>(z,  wt0, bd0, z1,
                                                                 BATCH, HID0, ZDIM);
  gemm_wmma<0><<<dim3(BATCH / 128, HID1 / 64), 256, 0, stream>>>(z1, wt1, bd1, z2,
                                                                 BATCH, HID1, HID0);
  gemm_wmma<1><<<dim3(BATCH / 128, DOUT / 64), 256, 0, stream>>>(z2, wt2, bd2, d_out,
                                                                 BATCH, DOUT, HID1);
}
